// AdjustableMarianAttention_81750407512574
// MI455X (gfx1250) — matrix operand order repaired and checked by reading, not compiled, not run on hardware
//
#include <hip/hip_runtime.h>
#include <hip/hip_bf16.h>
#include <math.h>

#define BB 4
#define HN 16
#define TT 1024
#define EDIM 1024
#define DH 64
#define BT (BB * TT)

typedef __attribute__((ext_vector_type(16))) __bf16        v16b;
typedef __attribute__((ext_vector_type(8)))  float         v8f;
typedef __attribute__((ext_vector_type(4)))  float         v4f;
typedef __attribute__((ext_vector_type(4)))  int           v4i;
typedef __attribute__((ext_vector_type(4)))  unsigned int  v4u;
typedef __attribute__((ext_vector_type(2)))  unsigned int  v2u;

union BFrag { v16b v; unsigned short u[16]; v4u q[2]; };

#if __has_builtin(__builtin_amdgcn_global_load_async_to_lds_b128)
#define HAVE_ASYNC_LDS 1
#else
#define HAVE_ASYNC_LDS 0
#endif

// copy one 16-byte chunk global->LDS (async path uses ASYNCcnt)
__device__ __forceinline__ void copy16_g2l(const unsigned short* g, unsigned short* l) {
#if HAVE_ASYNC_LDS
  unsigned short* gnc = const_cast<unsigned short*>(g);
  __builtin_amdgcn_global_load_async_to_lds_b128(
      (__attribute__((address_space(1))) v4i*)gnc,
      (__attribute__((address_space(3))) v4i*)l, 0, 0);
#else
  *(v4u*)l = *(const v4u*)g;
#endif
}

__device__ __forceinline__ void async_wait0() {
#if HAVE_ASYNC_LDS
#if __has_builtin(__builtin_amdgcn_s_wait_asynccnt)
  __builtin_amdgcn_s_wait_asynccnt(0);
#else
  asm volatile("s_wait_asynccnt 0" ::: "memory");
#endif
#endif
}

__device__ __forceinline__ unsigned short f2bf(float f) {
  union { float f; unsigned u; } x; x.f = f;
  unsigned r = x.u + 0x7FFFu + ((x.u >> 16) & 1u);
  return (unsigned short)(r >> 16);
}

__device__ __forceinline__ v8f wmma_bf16(const BFrag& a, const BFrag& b, v8f c) {
  return __builtin_amdgcn_wmma_f32_16x16x32_bf16(
      false, a.v, false, b.v, (short)0, c, false, false);
}

// LDS row strides (in ushorts): multiples of 8 (16B aligned rows) and
// stride/4 dwords == 4 mod 64 banks -> 16 distinct banks over 16 rows.
#define LDA 40
#define LDK 72
#define LDVT 40
#define LDP 40

// ---------------------------------------------------------------------------
// Kernel 1: fused Q/K/V projection.  C = (X @ W^T + bias) * scale.
// X f32 (BT x E), W f32 (ExE, rows = output features).  Output bf16 (B,H,T,D).
// grid.z selects Q/K/V.  256 thr = 8 waves (4x2), block tile 128x64.
// ---------------------------------------------------------------------------
__global__ __launch_bounds__(256) void qkv_proj_kernel(
    const float* __restrict__ X,
    const float* __restrict__ Wq, const float* __restrict__ bq,
    const float* __restrict__ Wk, const float* __restrict__ bk,
    const float* __restrict__ Wv, const float* __restrict__ bv,
    unsigned short* __restrict__ qo, unsigned short* __restrict__ ko,
    unsigned short* __restrict__ vo) {
  const int z = blockIdx.z;
  const float* W    = (z == 0) ? Wq : (z == 1) ? Wk : Wv;
  const float* bias = (z == 0) ? bq : (z == 1) ? bk : bv;
  unsigned short* out = (z == 0) ? qo : (z == 1) ? ko : vo;
  const float scale = (z == 0) ? 0.125f : 1.0f;  // D^-0.5 = 1/8

  const int m0 = blockIdx.x * 128;
  const int n0 = blockIdx.y * 64;

  __shared__ unsigned short lA[128 * LDA];
  __shared__ unsigned short lB[64 * LDA];

  const int tid  = threadIdx.x;
  const int wave = tid >> 5;
  const int lane = tid & 31;
  const int lh   = lane >> 4;
  const int lr   = lane & 15;
  const int wr   = wave & 3;
  const int wc   = wave >> 2;

  v8f acc[2][2];
#pragma unroll
  for (int i = 0; i < 2; ++i)
#pragma unroll
    for (int j = 0; j < 2; ++j)
#pragma unroll
      for (int e = 0; e < 8; ++e) acc[i][j][e] = 0.0f;

  for (int k0 = 0; k0 < EDIM; k0 += 32) {
    // stage A tile 128x32: float4 loads, packed bf16x4 stores
#pragma unroll
    for (int i = 0; i < 4; ++i) {
      int slot = tid + i * 256;           // 0..1023 float4 slots
      int r = slot >> 3, c4 = slot & 7;   // 8 float4 per row
      v4f x = *(const v4f*)&X[(size_t)(m0 + r) * EDIM + k0 + c4 * 4];
      v2u p;
      p[0] = (unsigned)f2bf(x[0]) | ((unsigned)f2bf(x[1]) << 16);
      p[1] = (unsigned)f2bf(x[2]) | ((unsigned)f2bf(x[3]) << 16);
      *(v2u*)&lA[r * LDA + c4 * 4] = p;
    }
    // stage B tile 64x32
#pragma unroll
    for (int i = 0; i < 2; ++i) {
      int slot = tid + i * 256;           // 0..511
      int r = slot >> 3, c4 = slot & 7;
      v4f x = *(const v4f*)&W[(size_t)(n0 + r) * EDIM + k0 + c4 * 4];
      v2u p;
      p[0] = (unsigned)f2bf(x[0]) | ((unsigned)f2bf(x[1]) << 16);
      p[1] = (unsigned)f2bf(x[2]) | ((unsigned)f2bf(x[3]) << 16);
      *(v2u*)&lB[r * LDA + c4 * 4] = p;
    }
    __syncthreads();

    BFrag a[2], b[2];
#pragma unroll
    for (int s = 0; s < 2; ++s) {
      int row = wr * 32 + s * 16 + lr;
      a[s].q[0] = *(const v4u*)&lA[row * LDA + 8 * lh];
      a[s].q[1] = *(const v4u*)&lA[row * LDA + 16 + 8 * lh];
    }
#pragma unroll
    for (int s = 0; s < 2; ++s) {
      int col = wc * 32 + s * 16 + lr;
      b[s].q[0] = *(const v4u*)&lB[col * LDA + 8 * lh];
      b[s].q[1] = *(const v4u*)&lB[col * LDA + 16 + 8 * lh];
    }
#pragma unroll
    for (int i = 0; i < 2; ++i)
#pragma unroll
      for (int j = 0; j < 2; ++j)
        acc[i][j] = wmma_bf16(a[i], b[j], acc[i][j]);
    __syncthreads();
  }

#pragma unroll
  for (int i = 0; i < 2; ++i) {
    int mbase = m0 + wr * 32 + i * 16 + 8 * lh;
#pragma unroll
    for (int j = 0; j < 2; ++j) {
      int n  = n0 + wc * 32 + j * 16 + lr;
      float bv_ = bias[n];
      int hh = n >> 6, d = n & 63;
#pragma unroll
      for (int r = 0; r < 8; ++r) {
        int gm = mbase + r;
        int bidx = gm >> 10;
        int t = gm & (TT - 1);
        float val = (acc[i][j][r] + bv_) * scale;
        out[(((size_t)(bidx * HN + hh) * TT) + t) * DH + d] = f2bf(val);
      }
    }
  }
}

// ---------------------------------------------------------------------------
// Kernel 2: fused attention + head disturbance (closed form):
//   out = (1-b)/a * sum_unmasked exp(s)*V + (b/n) * sum_masked V
// One wave per 16-query tile; 4 waves share LDS K (row-major) and V
// (transposed) tiles of 32 keys.  K tile staged with async-to-LDS copies.
// ---------------------------------------------------------------------------
__global__ __launch_bounds__(128) void attn_kernel(
    const unsigned short* __restrict__ qb,  // (B,H,T,D) bf16, pre-scaled
    const unsigned short* __restrict__ kb,  // (B,H,T,D) bf16
    const unsigned short* __restrict__ vb,  // (B,H,T,D) bf16
    const float* __restrict__ am,           // (B,1,T,T) f32
    const int* __restrict__ dmask,          // (B,H,T,T) int32
    unsigned short* __restrict__ ctx) {     // (B,T,E) bf16
  const int b  = blockIdx.z;
  const int hh = blockIdx.y;
  const int tid  = threadIdx.x;
  const int wave = tid >> 5;
  const int lane = tid & 31;
  const int lh   = lane >> 4;
  const int lr   = lane & 15;
  const int q0   = (blockIdx.x * 4 + wave) * 16;

  __shared__ unsigned short lK[32 * LDK];       // keys row-major (s, d)
  __shared__ unsigned short lVt[64 * LDVT];     // V transposed (d, s)
  __shared__ unsigned short lP[4][16 * LDP];
  __shared__ unsigned short lM[4][16 * LDP];

  const unsigned short* Qh = qb + (size_t)(b * HN + hh) * TT * DH;
  const unsigned short* Kh = kb + (size_t)(b * HN + hh) * TT * DH;
  const unsigned short* Vh = vb + (size_t)(b * HN + hh) * TT * DH;
  const float* amB = am + (size_t)b * TT * TT;
  const int* dmB = dmask + (size_t)(b * HN + hh) * TT * TT;

  // Q fragments (16x64 as two K=32 chunks), b128 global loads
  BFrag aq[2];
#pragma unroll
  for (int c = 0; c < 2; ++c) {
    aq[c].q[0] = *(const v4u*)&Qh[(size_t)(q0 + lr) * DH + c * 32 + 8 * lh];
    aq[c].q[1] = *(const v4u*)&Qh[(size_t)(q0 + lr) * DH + c * 32 + 16 + 8 * lh];
  }

  v8f accV[4], maccV[4];
#pragma unroll
  for (int t = 0; t < 4; ++t)
#pragma unroll
    for (int e = 0; e < 8; ++e) { accV[t][e] = 0.0f; maccV[t][e] = 0.0f; }

  float runmax[8], asum[8], ncnt[8];
#pragma unroll
  for (int r = 0; r < 8; ++r) { runmax[r] = -INFINITY; asum[r] = 0.0f; ncnt[r] = 0.0f; }

  for (int kb0 = 0; kb0 < TT; kb0 += 32) {
    // K tile 32x64: 256 chunks of 16B, async global->LDS
#pragma unroll
    for (int i = 0; i < 2; ++i) {
      int slot = tid + i * 128;           // 0..255
      int r = slot >> 3, c8 = slot & 7;
      copy16_g2l(&Kh[(size_t)(kb0 + r) * DH + c8 * 8], &lK[r * LDK + c8 * 8]);
    }
    // V tile staged transposed: b128 global load, scalar LDS scatter
#pragma unroll
    for (int i = 0; i < 2; ++i) {
      int slot = tid + i * 128;
      int r = slot >> 3, c8 = slot & 7;
      union { v4u q; unsigned short u[8]; } w;
      w.q = *(const v4u*)&Vh[(size_t)(kb0 + r) * DH + c8 * 8];
#pragma unroll
      for (int j = 0; j < 8; ++j)
        lVt[(c8 * 8 + j) * LDVT + r] = w.u[j];
    }
    if (kb0 + 32 < TT) {  // prefetch next block (global_prefetch_b8)
      int nr = tid & 31;
      __builtin_prefetch(Kh + (size_t)(kb0 + 32 + nr) * DH, 0, 1);
      __builtin_prefetch(Vh + (size_t)(kb0 + 32 + nr) * DH, 0, 1);
    }
    async_wait0();
    __syncthreads();

    // scores S = Q K^T for two 16-key subtiles
    float se[2][8], mv[2][8];
    float bmax[8];
#pragma unroll
    for (int r = 0; r < 8; ++r) bmax[r] = -INFINITY;

#pragma unroll
    for (int sj = 0; sj < 2; ++sj) {
      v8f sc;
#pragma unroll
      for (int e = 0; e < 8; ++e) sc[e] = 0.0f;
#pragma unroll
      for (int c = 0; c < 2; ++c) {
        BFrag bk_;
        bk_.q[0] = *(const v4u*)&lK[(sj * 16 + lr) * LDK + c * 32 + 8 * lh];
        bk_.q[1] = *(const v4u*)&lK[(sj * 16 + lr) * LDK + c * 32 + 16 + 8 * lh];
        sc = wmma_bf16(aq[c], bk_, sc);
      }
      int scol = kb0 + sj * 16 + lr;
#pragma unroll
      for (int r = 0; r < 8; ++r) {
        int qrow = q0 + 8 * lh + r;
        float s = sc[r] + amB[(size_t)qrow * TT + scol];
        int m = dmB[(size_t)qrow * TT + scol];
        se[sj][r] = s;
        mv[sj][r] = (float)m;
        if (!m) bmax[r] = fmaxf(bmax[r], s);
      }
    }

    // row max over the 16 lanes sharing each C row group
#pragma unroll
    for (int r = 0; r < 8; ++r) {
      float v_ = bmax[r];
      for (int off = 1; off < 16; off <<= 1)
        v_ = fmaxf(v_, __shfl_xor(v_, off, 32));
      bmax[r] = v_;
    }

    float sf[8], bsum[8], bcnt[8];
#pragma unroll
    for (int r = 0; r < 8; ++r) {
      float nm = fmaxf(runmax[r], bmax[r]);
      sf[r] = (nm == -INFINITY) ? 1.0f : __expf(runmax[r] - nm);
      runmax[r] = nm;
      bsum[r] = 0.0f; bcnt[r] = 0.0f;
    }

#pragma unroll
    for (int sj = 0; sj < 2; ++sj)
#pragma unroll
      for (int r = 0; r < 8; ++r) {
        float m = mv[sj][r];
        float pe = (m != 0.0f || runmax[r] == -INFINITY)
                       ? 0.0f
                       : __expf(se[sj][r] - runmax[r]);
        bsum[r] += pe;
        bcnt[r] += m;
        int row = 8 * lh + r, col = sj * 16 + lr;
        lP[wave][row * LDP + col] = f2bf(pe);
        lM[wave][row * LDP + col] = f2bf(m);
      }

#pragma unroll
    for (int r = 0; r < 8; ++r) {
      float s_ = bsum[r], c_ = bcnt[r];
      for (int off = 1; off < 16; off <<= 1) {
        s_ += __shfl_xor(s_, off, 32);
        c_ += __shfl_xor(c_, off, 32);
      }
      asum[r] = asum[r] * sf[r] + s_;
      ncnt[r] += c_;
    }

#pragma unroll
    for (int t = 0; t < 4; ++t)
#pragma unroll
      for (int r = 0; r < 8; ++r) accV[t][r] *= sf[r];

    // P and mask-matrix as A-operands (b128 LDS loads)
    BFrag ap, apm;
    ap.q[0]  = *(const v4u*)&lP[wave][lr * LDP + 8 * lh];
    ap.q[1]  = *(const v4u*)&lP[wave][lr * LDP + 16 + 8 * lh];
    apm.q[0] = *(const v4u*)&lM[wave][lr * LDP + 8 * lh];
    apm.q[1] = *(const v4u*)&lM[wave][lr * LDP + 16 + 8 * lh];

#pragma unroll
    for (int t = 0; t < 4; ++t) {
      BFrag bv_;  // V^T: rows = d, so contiguous along s
      bv_.q[0] = *(const v4u*)&lVt[(t * 16 + lr) * LDVT + 8 * lh];
      bv_.q[1] = *(const v4u*)&lVt[(t * 16 + lr) * LDVT + 16 + 8 * lh];
      accV[t]  = wmma_bf16(ap,  bv_, accV[t]);
      maccV[t] = wmma_bf16(apm, bv_, maccV[t]);
    }
    __syncthreads();
  }

  float w1[8], w2[8];
#pragma unroll
  for (int r = 0; r < 8; ++r) {
    float denom = (asum[r] > 0.0f) ? asum[r] : 1.0f;
    w1[r] = 0.5f / denom;
    w2[r] = 0.5f / fmaxf(ncnt[r], 1.0f);
  }
#pragma unroll
  for (int t = 0; t < 4; ++t)
#pragma unroll
    for (int r = 0; r < 8; ++r) {
      float val = accV[t][r] * w1[r] + maccV[t][r] * w2[r];
      int qrow = q0 + 8 * lh + r;
      ctx[((size_t)(b * TT + qrow)) * EDIM + hh * DH + t * 16 + lr] = f2bf(val);
    }
}

// ---------------------------------------------------------------------------
// Kernel 3: output projection.  out = ctx @ Wo^T + bo (ctx bf16, out f32).
// ---------------------------------------------------------------------------
__global__ __launch_bounds__(256) void out_proj_kernel(
    const unsigned short* __restrict__ Xb, const float* __restrict__ Wo,
    const float* __restrict__ bo, float* __restrict__ out) {
  const int m0 = blockIdx.x * 128;
  const int n0 = blockIdx.y * 64;

  __shared__ unsigned short lA[128 * LDA];
  __shared__ unsigned short lB[64 * LDA];

  const int tid  = threadIdx.x;
  const int wave = tid >> 5;
  const int lane = tid & 31;
  const int lh   = lane >> 4;
  const int lr   = lane & 15;
  const int wr   = wave & 3;
  const int wc   = wave >> 2;

  v8f acc[2][2];
#pragma unroll
  for (int i = 0; i < 2; ++i)
#pragma unroll
    for (int j = 0; j < 2; ++j)
#pragma unroll
      for (int e = 0; e < 8; ++e) acc[i][j][e] = 0.0f;

  for (int k0 = 0; k0 < EDIM; k0 += 32) {
    // A tile 128x32 bf16 passthrough: async global->LDS (512 chunks of 16B)
#pragma unroll
    for (int i = 0; i < 2; ++i) {
      int slot = tid + i * 256;           // 0..511
      int r = slot >> 2, c8 = slot & 3;   // 4 chunks of 8 ushorts per row
      copy16_g2l(&Xb[(size_t)(m0 + r) * EDIM + k0 + c8 * 8],
                 &lA[r * LDA + c8 * 8]);
    }
    // B tile 64x32 (f32 -> bf16)
#pragma unroll
    for (int i = 0; i < 2; ++i) {
      int slot = tid + i * 256;
      int r = slot >> 3, c4 = slot & 7;
      v4f x = *(const v4f*)&Wo[(size_t)(n0 + r) * EDIM + k0 + c4 * 4];
      v2u p;
      p[0] = (unsigned)f2bf(x[0]) | ((unsigned)f2bf(x[1]) << 16);
      p[1] = (unsigned)f2bf(x[2]) | ((unsigned)f2bf(x[3]) << 16);
      *(v2u*)&lB[r * LDA + c4 * 4] = p;
    }
    async_wait0();
    __syncthreads();

    BFrag a[2], b[2];
#pragma unroll
    for (int s = 0; s < 2; ++s) {
      int row = wr * 32 + s * 16 + lr;
      a[s].q[0] = *(const v4u*)&lA[row * LDA + 8 * lh];
      a[s].q[1] = *(const v4u*)&lA[row * LDA + 16 + 8 * lh];
    }
#pragma unroll
    for (int s = 0; s < 2; ++s) {
      int col = wc * 32 + s * 16 + lr;
      b[s].q[0] = *(const v4u*)&lB[col * LDA + 8 * lh];
      b[s].q[1] = *(const v4u*)&lB[col * LDA + 16 + 8 * lh];
    }
#pragma unroll
    for (int i = 0; i < 2; ++i)
#pragma unroll
      for (int j = 0; j < 2; ++j)
        acc[i][j] = wmma_bf16(a[i], b[j], acc[i][j]);
    __syncthreads();
  }

#pragma unroll
  for (int i = 0; i < 2; ++i) {
    int mbase = m0 + wr * 32 + i * 16 + 8 * lh;
#pragma unroll
    for (int j = 0; j < 2; ++j) {
      int n = n0 + wc * 32 + j * 16 + lr;
      float bv_ = bo[n];
#pragma unroll
      for (int r = 0; r < 8; ++r)
        out[(size_t)(mbase + r) * EDIM + n] = acc[i][j][r] + bv_;
    }
  }
}

extern "C" void kernel_launch(void* const* d_in, const int* in_sizes, int n_in,
                              void* d_out, int out_size, void* d_ws, size_t ws_size,
                              hipStream_t stream) {
  (void)in_sizes; (void)n_in; (void)out_size; (void)ws_size;
  const float* hs = (const float*)d_in[0];
  const float* am = (const float*)d_in[1];
  const int*   dm = (const int*)d_in[2];
  const float* Wq = (const float*)d_in[3];
  const float* bq = (const float*)d_in[4];
  const float* Wk = (const float*)d_in[5];
  const float* bk = (const float*)d_in[6];
  const float* Wv = (const float*)d_in[7];
  const float* bv = (const float*)d_in[8];
  const float* Wo = (const float*)d_in[9];
  const float* bo = (const float*)d_in[10];
  float* out = (float*)d_out;

  const size_t nElem = (size_t)BT * EDIM;  // 4M bf16 per buffer (8 MiB)
  unsigned short* q   = (unsigned short*)d_ws;
  unsigned short* k   = q + nElem;
  unsigned short* v   = k + nElem;
  unsigned short* ctx = v + nElem;  // total ws use: 32 MiB

  qkv_proj_kernel<<<dim3(BT / 128, EDIM / 64, 3), 256, 0, stream>>>(
      hs, Wq, bq, Wk, bk, Wv, bv, q, k, v);
  attn_kernel<<<dim3(TT / 64, HN, BB), 128, 0, stream>>>(q, k, v, am, dm, ctx);
  out_proj_kernel<<<dim3(BT / 128, EDIM / 64, 1), 256, 0, stream>>>(ctx, Wo, bo, out);
}
